// DynamicsModel_26388279066663
// MI455X (gfx1250) — matrix operand order repaired and checked by reading, not compiled, not run on hardware
//
#include <hip/hip_runtime.h>
#include <hip/hip_bf16.h>

typedef __attribute__((ext_vector_type(16))) _Float16 v16h;
typedef __attribute__((ext_vector_type(8)))  _Float16 v8h;
typedef __attribute__((ext_vector_type(8)))  float    v8f;

// Problem dims
constexpr int kB = 256, kT = 64, kH = 1024, kS = 256, kA = 64, kO = 1024;
constexpr int kSA = kS + kA;        // 320
constexpr int kHA = kH + kA;        // 1088
constexpr int kHO = kH + kO;        // 2048
constexpr int kG  = 3 * kH;         // 3072

// ---------------------------------------------------------------- activations
__device__ __forceinline__ float gelu_f(float x) {
  return 0.5f * x * (1.0f + erff(x * 0.70710678118654752440f));
}
__device__ __forceinline__ float sigm_f(float x) {
  return 1.0f / (1.0f + expf(-x));
}

// ----------------------------------------------- WMMA 32x32 register-blocked
// A: row-major [M x K] f16 (stride K). W: row-major [N x K] f16 (stride K),
// computing D = A @ W^T. Per K=32 step: 2 A-frags + 2 B-frags (8 x b128 loads)
// feed 4 v_wmma_f32_16x16x32_f16 -> 2.0 VMEM per WMMA.
// Fragment layouts per CDNA5 ISA 7.12.2 (wave32).
struct Acc2x2 { v8f a00, a01, a10, a11; };

__device__ __forceinline__ Acc2x2 gemm_tile32(const _Float16* __restrict__ A,
                                              const _Float16* __restrict__ W,
                                              int K, int m0, int n0, int lane) {
  const int hl = lane >> 4;   // half-wave select
  const int l  = lane & 15;
  const _Float16* ar0 = A + (size_t)(m0 + l) * K + hl * 8;
  const _Float16* ar1 = ar0 + (size_t)16 * K;
  const _Float16* wr0 = W + (size_t)(n0 + l) * K + hl * 8;
  const _Float16* wr1 = wr0 + (size_t)16 * K;
  Acc2x2 c;
  c.a00 = {}; c.a01 = {}; c.a10 = {}; c.a11 = {};
  for (int k = 0; k < K; k += 32) {
    union { v8h h[2]; v16h v; } a0, a1, b0, b1;
    a0.h[0] = *(const v8h*)(ar0 + k);      // K k..k+7 (hl=0) / k+8..k+15 (hl=1)
    a0.h[1] = *(const v8h*)(ar0 + k + 16); // K k+16..23     / k+24..31
    a1.h[0] = *(const v8h*)(ar1 + k);
    a1.h[1] = *(const v8h*)(ar1 + k + 16);
    b0.h[0] = *(const v8h*)(wr0 + k);      // same K split as A, W row n0+l
    b0.h[1] = *(const v8h*)(wr0 + k + 16);
    b1.h[0] = *(const v8h*)(wr1 + k);
    b1.h[1] = *(const v8h*)(wr1 + k + 16);
    c.a00 = __builtin_amdgcn_wmma_f32_16x16x32_f16(false, a0.v, false, b0.v,
                                                   (short)0, c.a00, false, false);
    c.a01 = __builtin_amdgcn_wmma_f32_16x16x32_f16(false, a0.v, false, b1.v,
                                                   (short)0, c.a01, false, false);
    c.a10 = __builtin_amdgcn_wmma_f32_16x16x32_f16(false, a1.v, false, b0.v,
                                                   (short)0, c.a10, false, false);
    c.a11 = __builtin_amdgcn_wmma_f32_16x16x32_f16(false, a1.v, false, b1.v,
                                                   (short)0, c.a11, false, false);
  }
  return c;
}

// ---------------------------------------------------------------- fp32 -> fp16
__global__ __launch_bounds__(256) void k_convert(const float* __restrict__ src,
                                                 _Float16* __restrict__ dst, int n) {
  int i = blockIdx.x * 256 + threadIdx.x;
  if (i < n) dst[i] = (_Float16)src[i];
}

// ------------------------------------------------------------------------ init
__global__ __launch_bounds__(256) void k_init(const float* __restrict__ prev_h,
                                              const float* __restrict__ prev_s,
                                              float* __restrict__ o_hid,
                                              float* __restrict__ o_pri,
                                              float* __restrict__ o_pos,
                                              float* __restrict__ hid,
                                              _Float16* __restrict__ h16,
                                              float* __restrict__ pst) {
  int i = blockIdx.x * 256 + threadIdx.x;
  if (i < kB * kH) {
    int b = i >> 10, n = i & (kH - 1);
    float v = prev_h[i];
    o_hid[(size_t)b * (kT + 1) * kH + n] = v;
    hid[i] = v;
    h16[i] = (_Float16)v;
  } else if (i < kB * kH + kB * kS) {
    int j = i - kB * kH;
    int b = j >> 8, s = j & (kS - 1);
    float v = prev_s[j];
    o_pri[(size_t)b * (kT + 1) * kS + s] = v;
    o_pos[(size_t)b * (kT + 1) * kS + s] = v;
    pst[j] = v;
  }
}

// ------------------------------------------------------------------- step prep
__global__ __launch_bounds__(256) void k_prep(const float* __restrict__ pst,
                                              const float* __restrict__ actions,
                                              const float* __restrict__ obs,
                                              const float* __restrict__ dones,
                                              _Float16* __restrict__ xsa,
                                              _Float16* __restrict__ ha_in,
                                              _Float16* __restrict__ ho_in,
                                              int t) {
  int i = blockIdx.x * 256 + threadIdx.x;
  const int n1 = kB * kSA, n2 = kB * kA, n3 = kB * kO;
  if (i < n1) {
    int b = i / kSA, c = i % kSA;
    if (c < kS) {
      float d = dones[b * kT + t];
      xsa[i] = (_Float16)(pst[b * kS + c] * (1.0f - d));
    } else {
      xsa[i] = (_Float16)actions[((size_t)b * kT + t) * kA + (c - kS)];
    }
  } else if (i < n1 + n2) {
    int j = i - n1;
    int b = j >> 6, c = j & (kA - 1);
    ha_in[b * kHA + kH + c] = (_Float16)actions[((size_t)b * kT + t) * kA + c];
  } else if (i < n1 + n2 + n3) {
    int j = i - n1 - n2;
    int b = j >> 10, c = j & (kO - 1);
    ho_in[b * kHO + kH + c] = (_Float16)obs[((size_t)b * kT + t) * kO + c];
  }
}

// ------------------------------------------------------------- sa = gelu GEMM
__global__ __launch_bounds__(256) void k_sa(const _Float16* __restrict__ xsa,
                                            const _Float16* __restrict__ Wsa,
                                            const float* __restrict__ bsa,
                                            _Float16* __restrict__ sa) {
  int gw = (blockIdx.x * 256 + threadIdx.x) >> 5;   // 8 x 32 = 256 waves
  int lane = threadIdx.x & 31;
  int m0 = (gw >> 5) << 5, n0 = (gw & 31) << 5;
  Acc2x2 c = gemm_tile32(xsa, Wsa, kSA, m0, n0, lane);
  int hl = lane >> 4, l = lane & 15;
  auto emit = [&](const v8f& acc, int mi, int ni) {
    int col = n0 + ni * 16 + l;
    float bv = bsa[col];
#pragma unroll
    for (int i = 0; i < 8; ++i) {
      int row = m0 + mi * 16 + hl * 8 + i;
      sa[row * kH + col] = (_Float16)gelu_f(acc[i] + bv);
    }
  };
  emit(c.a00, 0, 0); emit(c.a01, 0, 1); emit(c.a10, 1, 0); emit(c.a11, 1, 1);
}

// ------------------------------------------- gi = sa@Wih^T ; gh = h@Whh^T (f32)
__global__ __launch_bounds__(256) void k_gates(const _Float16* __restrict__ sa,
                                               const _Float16* __restrict__ h16,
                                               const _Float16* __restrict__ Wih,
                                               const _Float16* __restrict__ Whh,
                                               const float* __restrict__ bih,
                                               const float* __restrict__ bhh,
                                               float* __restrict__ gi,
                                               float* __restrict__ gh) {
  int gw = (blockIdx.x * 256 + threadIdx.x) >> 5;   // 2 x 8 x 96 = 1536 waves
  int lane = threadIdx.x & 31;
  bool isGh = gw >= (8 * 96);
  int w = isGh ? gw - 8 * 96 : gw;
  int m0 = (w / 96) << 5, n0 = (w % 96) << 5;
  const _Float16* A = isGh ? h16 : sa;
  const _Float16* W = isGh ? Whh : Wih;
  const float* bias = isGh ? bhh : bih;
  float* outp       = isGh ? gh : gi;
  Acc2x2 c = gemm_tile32(A, W, kH, m0, n0, lane);
  int hl = lane >> 4, l = lane & 15;
  auto emit = [&](const v8f& acc, int mi, int ni) {
    int col = n0 + ni * 16 + l;
    float bv = bias[col];
#pragma unroll
    for (int i = 0; i < 8; ++i) {
      int row = m0 + mi * 16 + hl * 8 + i;
      outp[(size_t)row * kG + col] = acc[i] + bv;
    }
  };
  emit(c.a00, 0, 0); emit(c.a01, 0, 1); emit(c.a10, 1, 0); emit(c.a11, 1, 1);
}

// --------------------------------------------------------------- GRU combine
__global__ __launch_bounds__(256) void k_gru(const float* __restrict__ gi,
                                             const float* __restrict__ gh,
                                             float* __restrict__ hid,
                                             _Float16* __restrict__ h16,
                                             _Float16* __restrict__ ha_in,
                                             _Float16* __restrict__ ho_in,
                                             float* __restrict__ o_hid, int t) {
  int i = blockIdx.x * 256 + threadIdx.x;
  if (i >= kB * kH) return;
  int b = i >> 10, n = i & (kH - 1);
  size_t r = (size_t)b * kG;
  float rg = sigm_f(gi[r + n] + gh[r + n]);
  float zg = sigm_f(gi[r + kH + n] + gh[r + kH + n]);
  float ng = tanhf(gi[r + 2 * kH + n] + rg * gh[r + 2 * kH + n]);
  float hnew = (1.0f - zg) * ng + zg * hid[i];
  hid[i] = hnew;
  h16[i] = (_Float16)hnew;
  ha_in[b * kHA + n] = (_Float16)hnew;
  ho_in[b * kHO + n] = (_Float16)hnew;
  o_hid[(size_t)b * (kT + 1) * kH + (size_t)(t + 1) * kH + n] = hnew;
}

// ----------------------------------- ha = gelu([h|a]@Wha^T) ; ho = gelu([h|o]@..)
__global__ __launch_bounds__(256) void k_haho(const _Float16* __restrict__ ha_in,
                                              const _Float16* __restrict__ ho_in,
                                              const _Float16* __restrict__ Wha,
                                              const _Float16* __restrict__ Who,
                                              const float* __restrict__ bha,
                                              const float* __restrict__ bho,
                                              _Float16* __restrict__ ha,
                                              _Float16* __restrict__ ho) {
  int gw = (blockIdx.x * 256 + threadIdx.x) >> 5;   // 2 x 8 x 32 = 512 waves
  int lane = threadIdx.x & 31;
  bool isHo = gw >= (8 * 32);
  int w = isHo ? gw - 8 * 32 : gw;
  int m0 = (w >> 5) << 5, n0 = (w & 31) << 5;
  Acc2x2 c = isHo ? gemm_tile32(ho_in, Who, kHO, m0, n0, lane)
                  : gemm_tile32(ha_in, Wha, kHA, m0, n0, lane);
  const float* bias = isHo ? bho : bha;
  _Float16* outp    = isHo ? ho : ha;
  int hl = lane >> 4, l = lane & 15;
  auto emit = [&](const v8f& acc, int mi, int ni) {
    int col = n0 + ni * 16 + l;
    float bv = bias[col];
#pragma unroll
    for (int i = 0; i < 8; ++i) {
      int row = m0 + mi * 16 + hl * 8 + i;
      outp[row * kH + col] = (_Float16)gelu_f(acc[i] + bv);
    }
  };
  emit(c.a00, 0, 0); emit(c.a01, 0, 1); emit(c.a10, 1, 0); emit(c.a11, 1, 1);
}

// ---------------------------------------------- pp = ha@Wpr^T ; qp = ho@Wpo^T
__global__ __launch_bounds__(256) void k_ppqp(const _Float16* __restrict__ ha,
                                              const _Float16* __restrict__ ho,
                                              const _Float16* __restrict__ Wpr,
                                              const _Float16* __restrict__ Wpo,
                                              const float* __restrict__ bpr,
                                              const float* __restrict__ bpo,
                                              float* __restrict__ pp,
                                              float* __restrict__ qp) {
  int gw = (blockIdx.x * 256 + threadIdx.x) >> 5;   // 2 x 8 x 16 = 256 waves
  int lane = threadIdx.x & 31;
  bool isQ = gw >= (8 * 16);
  int w = isQ ? gw - 8 * 16 : gw;
  int m0 = (w >> 4) << 5, n0 = (w & 15) << 5;
  Acc2x2 c = isQ ? gemm_tile32(ho, Wpo, kH, m0, n0, lane)
                 : gemm_tile32(ha, Wpr, kH, m0, n0, lane);
  const float* bias = isQ ? bpo : bpr;
  float* outp       = isQ ? qp : pp;
  int hl = lane >> 4, l = lane & 15;
  auto emit = [&](const v8f& acc, int mi, int ni) {
    int col = n0 + ni * 16 + l;
    float bv = bias[col];
#pragma unroll
    for (int i = 0; i < 8; ++i) {
      int row = m0 + mi * 16 + hl * 8 + i;
      outp[row * (2 * kS) + col] = acc[i] + bv;
    }
  };
  emit(c.a00, 0, 0); emit(c.a01, 0, 1); emit(c.a10, 1, 0); emit(c.a11, 1, 1);
}

// ----------------------------------------------------- reparameterize + write
// exp(softplus(x)) == 1 + exp(x)
__global__ __launch_bounds__(256) void k_sample(const float* __restrict__ pp,
                                                const float* __restrict__ qp,
                                                const float* __restrict__ eps_pr,
                                                const float* __restrict__ eps_po,
                                                float* __restrict__ pst,
                                                float* __restrict__ o_pri,
                                                float* __restrict__ o_pos,
                                                float* __restrict__ o_pm,
                                                float* __restrict__ o_plv,
                                                float* __restrict__ o_qm,
                                                float* __restrict__ o_qlv, int t) {
  int i = blockIdx.x * 256 + threadIdx.x;
  if (i >= kB * kS) return;
  int b = i >> 8, s = i & (kS - 1);
  size_t bt = ((size_t)b * kT + t) * kS + s;
  size_t r = (size_t)b * (2 * kS);
  float pm = pp[r + s], plv = pp[r + kS + s];
  float pr = pm + (1.0f + expf(plv)) * eps_pr[bt];
  float qm = qp[r + s], qlv = qp[r + kS + s];
  float po = qm + (1.0f + expf(qlv)) * eps_po[bt];
  o_pm[bt] = pm;  o_plv[bt] = plv;
  o_qm[bt] = qm;  o_qlv[bt] = qlv;
  size_t seq = (size_t)b * (kT + 1) * kS + (size_t)(t + 1) * kS + s;
  o_pri[seq] = pr;
  o_pos[seq] = po;
  pst[i] = po;
}

// ------------------------------------------------------------------------ host
extern "C" void kernel_launch(void* const* d_in, const int* in_sizes, int n_in,
                              void* d_out, int out_size, void* d_ws, size_t ws_size,
                              hipStream_t stream) {
  (void)in_sizes; (void)n_in; (void)out_size; (void)ws_size;
  const float* prev_hidden = (const float*)d_in[0];
  const float* prev_state  = (const float*)d_in[1];
  const float* actions     = (const float*)d_in[2];
  const float* obs         = (const float*)d_in[3];
  const float* dones       = (const float*)d_in[4];
  const float* eps_prior   = (const float*)d_in[5];
  const float* eps_post    = (const float*)d_in[6];
  const float* W_sa   = (const float*)d_in[7];   const float* b_sa   = (const float*)d_in[8];
  const float* W_ih   = (const float*)d_in[9];   const float* W_hh   = (const float*)d_in[10];
  const float* b_ih   = (const float*)d_in[11];  const float* b_hh   = (const float*)d_in[12];
  const float* W_ha   = (const float*)d_in[13];  const float* b_ha   = (const float*)d_in[14];
  const float* W_pr   = (const float*)d_in[15];  const float* b_pr   = (const float*)d_in[16];
  const float* W_ho   = (const float*)d_in[17];  const float* b_ho   = (const float*)d_in[18];
  const float* W_po   = (const float*)d_in[19];  const float* b_po   = (const float*)d_in[20];

  // ---- workspace carve-up: f16 region first, then f32 region
  _Float16* f16 = (_Float16*)d_ws;
  size_t p = 0;
  _Float16* Wsa16 = f16 + p; p += (size_t)kH * kSA;
  _Float16* Wih16 = f16 + p; p += (size_t)kG * kH;
  _Float16* Whh16 = f16 + p; p += (size_t)kG * kH;
  _Float16* Wha16 = f16 + p; p += (size_t)kH * kHA;
  _Float16* Wpr16 = f16 + p; p += (size_t)(2 * kS) * kH;
  _Float16* Who16 = f16 + p; p += (size_t)kH * kHO;
  _Float16* Wpo16 = f16 + p; p += (size_t)(2 * kS) * kH;
  _Float16* xsa   = f16 + p; p += (size_t)kB * kSA;
  _Float16* sa    = f16 + p; p += (size_t)kB * kH;
  _Float16* h16   = f16 + p; p += (size_t)kB * kH;
  _Float16* ha_in = f16 + p; p += (size_t)kB * kHA;
  _Float16* ha    = f16 + p; p += (size_t)kB * kH;
  _Float16* ho_in = f16 + p; p += (size_t)kB * kHO;
  _Float16* ho    = f16 + p; p += (size_t)kB * kH;
  float* f32 = (float*)(f16 + p);
  size_t q = 0;
  float* hid = f32 + q; q += (size_t)kB * kH;
  float* pst = f32 + q; q += (size_t)kB * kS;
  float* gi  = f32 + q; q += (size_t)kB * kG;
  float* gh  = f32 + q; q += (size_t)kB * kG;
  float* pp  = f32 + q; q += (size_t)kB * (2 * kS);
  float* qp  = f32 + q; q += (size_t)kB * (2 * kS);

  // ---- output carve-up (reference tuple, flat, in return order)
  float* out = (float*)d_out;
  float* o_hid = out;
  float* o_pri = o_hid + (size_t)kB * (kT + 1) * kH;
  float* o_pos = o_pri + (size_t)kB * (kT + 1) * kS;
  float* o_pm  = o_pos + (size_t)kB * (kT + 1) * kS;
  float* o_plv = o_pm  + (size_t)kB * kT * kS;
  float* o_qm  = o_plv + (size_t)kB * kT * kS;
  float* o_qlv = o_qm  + (size_t)kB * kT * kS;

  auto cvt = [&](const float* s, _Float16* d, size_t n) {
    k_convert<<<(unsigned)((n + 255) / 256), 256, 0, stream>>>(s, d, (int)n);
  };
  cvt(W_sa, Wsa16, (size_t)kH * kSA);
  cvt(W_ih, Wih16, (size_t)kG * kH);
  cvt(W_hh, Whh16, (size_t)kG * kH);
  cvt(W_ha, Wha16, (size_t)kH * kHA);
  cvt(W_pr, Wpr16, (size_t)(2 * kS) * kH);
  cvt(W_ho, Who16, (size_t)kH * kHO);
  cvt(W_po, Wpo16, (size_t)(2 * kS) * kH);

  {
    int n = kB * kH + kB * kS;
    k_init<<<(n + 255) / 256, 256, 0, stream>>>(prev_hidden, prev_state,
                                                o_hid, o_pri, o_pos, hid, h16, pst);
  }

  for (int t = 0; t < kT; ++t) {
    {
      int n = kB * kSA + kB * kA + kB * kO;
      k_prep<<<(n + 255) / 256, 256, 0, stream>>>(pst, actions, obs, dones,
                                                  xsa, ha_in, ho_in, t);
    }
    k_sa   <<<(8 * 32)      / 8, 256, 0, stream>>>(xsa, Wsa16, b_sa, sa);
    k_gates<<<(2 * 8 * 96)  / 8, 256, 0, stream>>>(sa, h16, Wih16, Whh16,
                                                   b_ih, b_hh, gi, gh);
    k_gru  <<<(kB * kH) / 256, 256, 0, stream>>>(gi, gh, hid, h16, ha_in, ho_in,
                                                 o_hid, t);
    k_haho <<<(2 * 8 * 32)  / 8, 256, 0, stream>>>(ha_in, ho_in, Wha16, Who16,
                                                   b_ha, b_ho, ha, ho);
    k_ppqp <<<(2 * 8 * 16)  / 8, 256, 0, stream>>>(ha, ho, Wpr16, Wpo16,
                                                   b_pr, b_po, pp, qp);
    k_sample<<<(kB * kS) / 256, 256, 0, stream>>>(pp, qp, eps_prior, eps_post,
                                                  pst, o_pri, o_pos,
                                                  o_pm, o_plv, o_qm, o_qlv, t);
  }
}
